// NN_57844619543085
// MI455X (gfx1250) — matrix operand order repaired and checked by reading, not compiled, not run on hardware
//
#include <hip/hip_runtime.h>

typedef __attribute__((ext_vector_type(16))) _Float16 v16h;
typedef __attribute__((ext_vector_type(8)))  _Float16 v8h;
typedef __attribute__((ext_vector_type(8)))  float    v8f;
typedef __attribute__((ext_vector_type(4)))  float    v4f;

namespace {
constexpr int KDIM          = 512;
constexpr int NDIM          = 64;
constexpr int BATCH         = 65536;
constexpr int BLOCK_THREADS = 256;          // 8 waves (wave32)
constexpr int ROWS_PER_BLOCK = 128;         // 8 waves x 16 rows
constexpr int LDS_STRIDE    = KDIM + 8;     // pad: avoid 16-way bank conflicts on B reads
}

// logits[b, o] = sum_k x[b, k] * W[k, o]
// f32 inputs split into (hi, lo) f16 pairs; D accumulates in f32 via
// V_WMMA_F32_16X16X32_F16: Ahi*Bhi + Ahi*Blo + Alo*Bhi  (lo*lo ~ 2^-22, dropped)
__global__ __launch_bounds__(BLOCK_THREADS)
void neat_logits_wmma(const float* __restrict__ x,
                      const float* __restrict__ w,
                      float* __restrict__ out) {
  __shared__ _Float16 sWhi[NDIM * LDS_STRIDE];
  __shared__ _Float16 sWlo[NDIM * LDS_STRIDE];

  const int tid = threadIdx.x;

  // --- one-time per block: split W (k-major [512][64]) into f16 hi/lo, transposed [n][k]
  for (int idx = tid; idx < KDIM * NDIM; idx += BLOCK_THREADS) {
    const int k = idx >> 6;              // row (K)
    const int n = idx & (NDIM - 1);      // col (N), fastest in memory
    const float wv = w[idx];
    const _Float16 hi = (_Float16)wv;
    const _Float16 lo = (_Float16)(wv - (float)hi);
    sWhi[n * LDS_STRIDE + k] = hi;
    sWlo[n * LDS_STRIDE + k] = lo;
  }
  __syncthreads();

  const int lane    = tid & 31;
  const int nl      = lane & 15;         // N-lane / M-lane inside tile
  const int khalf   = lane >> 4;         // lanes 16..31 hold the other K half
  const int waveRow = blockIdx.x * ROWS_PER_BLOCK + (tid >> 5) * 16;

  const float* xrow = x + (size_t)(waveRow + nl) * KDIM;

  v8f acc[4] = {};                       // 16x64 f32 output tile per wave

  for (int kb = 0; kb < KDIM; kb += 32) {
    // A fragment (16-bit A 16x32): lane<16 holds K {kb..kb+7, kb+16..kb+23},
    // lane>=16 holds K {kb+8..kb+15, kb+24..kb+31}, row M = nl.
    const int k0 = kb + khalf * 8;
    v4f a0 = __builtin_nontemporal_load((const v4f*)(xrow + k0));
    v4f a1 = __builtin_nontemporal_load((const v4f*)(xrow + k0 + 4));
    v4f a2 = __builtin_nontemporal_load((const v4f*)(xrow + k0 + 16));
    v4f a3 = __builtin_nontemporal_load((const v4f*)(xrow + k0 + 20));

    float xs[16];
#pragma unroll
    for (int j = 0; j < 4; ++j) {
      xs[j]      = a0[j];
      xs[4 + j]  = a1[j];
      xs[8 + j]  = a2[j];
      xs[12 + j] = a3[j];
    }
    v16h ahi, alo;
#pragma unroll
    for (int j = 0; j < 16; ++j) {
      const float v = xs[j];
      const _Float16 h = (_Float16)v;
      ahi[j] = h;
      alo[j] = (_Float16)(v - (float)h);   // exact residual rounded to f16
    }

    // B fragment (16-bit B 32x16): lanes 0-15 hold K {kb..kb+7, kb+16..kb+23} for N=nl,
    // lanes 16-31 hold K {kb+8..kb+15, kb+24..kb+31} — matches the A fragment.
    const int kB = kb + khalf * 8;
#pragma unroll
    for (int nt = 0; nt < 4; ++nt) {
      const _Float16* bhp = &sWhi[(nt * 16 + nl) * LDS_STRIDE + kB];
      const _Float16* blp = &sWlo[(nt * 16 + nl) * LDS_STRIDE + kB];
      v8h bh0 = *(const v8h*)bhp;
      v8h bh1 = *(const v8h*)(bhp + 16);
      v8h bl0 = *(const v8h*)blp;
      v8h bl1 = *(const v8h*)(blp + 16);
      v16h bhi = __builtin_shufflevector(bh0, bh1, 0,1,2,3,4,5,6,7,8,9,10,11,12,13,14,15);
      v16h blo = __builtin_shufflevector(bl0, bl1, 0,1,2,3,4,5,6,7,8,9,10,11,12,13,14,15);

      acc[nt] = __builtin_amdgcn_wmma_f32_16x16x32_f16(false, ahi, false, bhi,
                                                       (short)0, acc[nt], false, false);
      acc[nt] = __builtin_amdgcn_wmma_f32_16x16x32_f16(false, ahi, false, blo,
                                                       (short)0, acc[nt], false, false);
      acc[nt] = __builtin_amdgcn_wmma_f32_16x16x32_f16(false, alo, false, bhi,
                                                       (short)0, acc[nt], false, false);
    }
  }

  // C/D layout: VGPR i holds M = i (lanes 0-15) / M = 8+i (lanes 16-31), N = nl.
  float* obase = out + (size_t)(waveRow + khalf * 8) * NDIM + nl;
#pragma unroll
  for (int i = 0; i < 8; ++i) {
#pragma unroll
    for (int nt = 0; nt < 4; ++nt) {
      __builtin_nontemporal_store(acc[nt][i], obase + (size_t)i * NDIM + nt * 16);
    }
  }
}

extern "C" void kernel_launch(void* const* d_in, const int* in_sizes, int n_in,
                              void* d_out, int out_size, void* d_ws, size_t ws_size,
                              hipStream_t stream) {
  (void)in_sizes; (void)n_in; (void)out_size; (void)d_ws; (void)ws_size;
  const float* x = (const float*)d_in[0];
  const float* w = (const float*)d_in[1];
  float* out = (float*)d_out;
  dim3 grid(BATCH / ROWS_PER_BLOCK);
  dim3 block(BLOCK_THREADS);
  hipLaunchKernelGGL(neat_logits_wmma, grid, block, 0, stream, x, w, out);
}
